// PairRBFModel_52158082842743
// MI455X (gfx1250) — compile-verified, matrix operand order checked by reading, not run on hardware
//
#include <hip/hip_runtime.h>
#include <hip/hip_bf16.h>

// ---------------- model constants ----------------
#define NATOM   1536
#define NSPEC   8
#define HID     64
#define NC      32
#define CUTR    5.0f
#define GAMMA_R 4.0f
#define LOCAL_IN 289           // NC*NSPEC + HID/2 + 1
#define PI_F    3.14159265358979f

constexpr int NPAIR = NATOM * (NATOM - 1) / 2;   // 1,178,880
constexpr int NTILE = NPAIR / 16;                // 73,680 (exact)

typedef __attribute__((ext_vector_type(16))) _Float16 v16h;
typedef __attribute__((ext_vector_type(8)))  float    v8f;

// LDS pitches (in halfs) chosen odd-in-dwords to avoid bank conflicts
#define W1T_STRIDE 98   // 96 + 2
#define W2T_STRIDE 66   // 64 + 2
#define H1_STRIDE  66   // 64 + 2

// ---------------- helpers ----------------

// pairs with first index < i
__device__ __forceinline__ int pair_offset(int i) {
    return i * NATOM - ((i * (i + 1)) >> 1);
}

// invert linear upper-triangular pair index p -> (i, j), i < j
__device__ __forceinline__ void pair_ij(int p, int& io, int& jo) {
    float b = 2.0f * (float)NATOM - 1.0f;               // 3071
    float disc = b * b - 8.0f * (float)p;
    int i = (int)((b - sqrtf(fmaxf(disc, 0.0f))) * 0.5f);
    if (i > NATOM - 2) i = NATOM - 2;
    if (i < 0) i = 0;
    while (pair_offset(i) > p) --i;
    while (i < NATOM - 2 && pair_offset(i + 1) <= p) ++i;
    io = i;
    jo = p - pair_offset(i) + i + 1;
}

__device__ __forceinline__ float cutoff_fn(float d) {
    float x = d * (1.0f / CUTR);
    x = fminf(fmaxf(x, 0.0f), 1.0f);
    return (d < CUTR) ? 0.5f * (__cosf(PI_F * x) + 1.0f) : 0.0f;
}

__device__ __forceinline__ float silu(float x) {
    return x / (1.0f + __expf(-x));
}

__device__ __forceinline__ void wait_ds() {
#if defined(__HIP_DEVICE_COMPILE__)
    asm volatile("s_wait_dscnt 0" ::: "memory");
#endif
}

// Load a 16x32 K-chunk fragment (A or B share the same per-lane K pattern):
// register r holds K = kb, kb+1 with kb = (r/4)*16 + half*8 + (r%4)*2
__device__ __forceinline__ v16h load_frag_k(const _Float16* p, int hf) {
    v16h b;
#pragma unroll
    for (int r = 0; r < 8; ++r) {
        int kb = ((r >> 2) << 4) + (hf << 3) + ((r & 3) << 1);
        b[2 * r]     = p[kb];
        b[2 * r + 1] = p[kb + 1];
    }
    return b;
}

__device__ __forceinline__ v8f wmma16(v16h a, v16h b, v8f c) {
    return __builtin_amdgcn_wmma_f32_16x16x32_f16(
        /*neg_a=*/false, a, /*neg_b=*/false, b,
        /*c_mod=*/(short)0, c, /*reuse_a=*/false, /*reuse_b=*/false);
}

// ---------------- kernel 1: energy init (atom-bias sum) ----------------
__global__ void init_kernel(const int* __restrict__ species,
                            const float* __restrict__ atom_bias,
                            float* __restrict__ out) {
    __shared__ float red[256];
    int tid = threadIdx.x;
    float s = 0.0f;
    for (int k = tid; k < NATOM; k += 256) s += atom_bias[species[k]];
    red[tid] = s;
    __syncthreads();
    for (int off = 128; off > 0; off >>= 1) {
        if (tid < off) red[tid] += red[tid + off];
        __syncthreads();
    }
    if (tid == 0) out[0] = red[0];
}

// ---------------- kernel 2: WMMA pair-energy MLP ----------------
__global__ __launch_bounds__(256) void pair_energy_kernel(
    const int* __restrict__ species, const float* __restrict__ pos,
    const float* __restrict__ embed,
    const float* __restrict__ W1, const float* __restrict__ b1,
    const float* __restrict__ W2, const float* __restrict__ b2,
    const float* __restrict__ W3, const float* __restrict__ b3,
    float* __restrict__ out) {
    __shared__ _Float16 sW1t[64 * W1T_STRIDE];     // [col][k], k<96
    __shared__ _Float16 sW2t[64 * W2T_STRIDE];     // [col][k], k<64
    __shared__ _Float16 sEmb[NSPEC * 32];
    __shared__ float sB1[64], sB2[64], sW3v[64];
    __shared__ _Float16 sH1[8 * 16 * H1_STRIDE];   // per-wave 16x64 staging

    const int tid = threadIdx.x;
    for (int idx = tid; idx < 96 * 64; idx += 256) {
        int k = idx >> 6, c = idx & 63;
        sW1t[c * W1T_STRIDE + k] = (_Float16)W1[idx];
    }
    for (int idx = tid; idx < 64 * 64; idx += 256) {
        int k = idx >> 6, c = idx & 63;
        sW2t[c * W2T_STRIDE + k] = (_Float16)W2[idx];
    }
    for (int idx = tid; idx < NSPEC * 32; idx += 256)
        sEmb[idx] = (_Float16)embed[idx];
    if (tid < 64) { sB1[tid] = b1[tid]; sB2[tid] = b2[tid]; sW3v[tid] = W3[tid]; }
    __syncthreads();

    const int lane = tid & 31;
    const int wave = tid >> 5;
    const int m  = lane & 15;     // pair row within tile / output col
    const int hf = lane >> 4;     // half selector
    const float b3v = b3[0];
    _Float16* myH1 = sH1 + wave * (16 * H1_STRIDE);

    // ---- hoist layer-1 weight B-fragments into registers (96 VGPRs) ----
    v16h B1f[3][4];
#pragma unroll
    for (int nt = 0; nt < 4; ++nt) {
        const _Float16* wc1 = sW1t + (nt * 16 + m) * W1T_STRIDE;
#pragma unroll
        for (int c = 0; c < 3; ++c) B1f[c][nt] = load_frag_k(wc1 + 32 * c, hf);
    }
    const float bias1[4] = { sB1[m], sB1[16 + m], sB1[32 + m], sB1[48 + m] };
    const float bias2[4] = { sB2[m], sB2[16 + m], sB2[32 + m], sB2[48 + m] };
    const float w3c[4]   = { sW3v[m], sW3v[16 + m], sW3v[32 + m], sW3v[48 + m] };
    const _Float16* wc2base = sW2t + m * W2T_STRIDE;

    float eacc = 0.0f;
    const int gw = blockIdx.x * 8 + wave;
    const int nw = gridDim.x * 8;

    for (int tile = gw; tile < NTILE; tile += nw) {
        // ---- per-pair geometry / features (lane owns pair m of the tile) ----
        int p = tile * 16 + m;
        int i, j; pair_ij(p, i, j);
        float dx = pos[3 * i + 0] - pos[3 * j + 0];
        float dy = pos[3 * i + 1] - pos[3 * j + 1];
        float dz = pos[3 * i + 2] - pos[3 * j + 2];
        float rij = sqrtf(dx * dx + dy * dy + dz * dz + 1e-12f);
        float cut = cutoff_fn(rij);
        int si = species[i], sj = species[j];

        // Build A fragments directly in WMMA layout
        v16h a0, a1, a2;
#pragma unroll
        for (int r = 0; r < 8; ++r) {
            int kb = ((r >> 2) << 4) + (hf << 3) + ((r & 3) << 1);
            float c0 = rij - (float)kb * (CUTR / 31.0f);
            float c1 = rij - (float)(kb + 1) * (CUTR / 31.0f);
            a0[2 * r]     = (_Float16)__expf(-GAMMA_R * c0 * c0);
            a0[2 * r + 1] = (_Float16)__expf(-GAMMA_R * c1 * c1);
            _Float16 e0 = sEmb[si * 32 + kb], e1 = sEmb[si * 32 + kb + 1];
            _Float16 f0 = sEmb[sj * 32 + kb], f1 = sEmb[sj * 32 + kb + 1];
            a1[2 * r] = e0 + f0; a1[2 * r + 1] = e1 + f1;
            a2[2 * r] = e0 * f0; a2[2 * r + 1] = e1 * f1;
        }

        // ---- layer 1: [16,96] x [96,64], 3 K-chunks x 4 N-tiles ----
        v8f acc[4];
#pragma unroll
        for (int nt = 0; nt < 4; ++nt) {
#pragma unroll
            for (int v = 0; v < 8; ++v) acc[nt][v] = bias1[nt];
        }
#pragma unroll
        for (int nt = 0; nt < 4; ++nt) {
            acc[nt] = wmma16(a0, B1f[0][nt], acc[nt]);
            acc[nt] = wmma16(a1, B1f[1][nt], acc[nt]);
            acc[nt] = wmma16(a2, B1f[2][nt], acc[nt]);
        }

        // ---- prefetch layer-2 B fragments (latency hidden under silu/store) --
        v16h B2f[2][4];
#pragma unroll
        for (int nt = 0; nt < 4; ++nt) {
#pragma unroll
            for (int c = 0; c < 2; ++c)
                B2f[c][nt] = load_frag_k(wc2base + nt * 16 * W2T_STRIDE + 32 * c, hf);
        }

        // ---- silu + relayout C/D -> A via per-wave LDS staging ----
#pragma unroll
        for (int nt = 0; nt < 4; ++nt)
#pragma unroll
            for (int v = 0; v < 8; ++v)
                myH1[(v + 8 * hf) * H1_STRIDE + nt * 16 + m] =
                    (_Float16)silu(acc[nt][v]);
        wait_ds();

        // ---- layer 2: [16,64] x [64,64], 2 K-chunks x 4 N-tiles ----
        v16h ax0 = load_frag_k(myH1 + m * H1_STRIDE +  0, hf);
        v16h ax1 = load_frag_k(myH1 + m * H1_STRIDE + 32, hf);
        v8f acc2[4];
#pragma unroll
        for (int nt = 0; nt < 4; ++nt) {
#pragma unroll
            for (int v = 0; v < 8; ++v) acc2[nt][v] = bias2[nt];
        }
#pragma unroll
        for (int nt = 0; nt < 4; ++nt) {
            acc2[nt] = wmma16(ax0, B2f[0][nt], acc2[nt]);
            acc2[nt] = wmma16(ax1, B2f[1][nt], acc2[nt]);
        }

        // ---- layer 3: silu(h2) . W3, per-row reduction over 16 lanes ----
        float part[8];
#pragma unroll
        for (int v = 0; v < 8; ++v) part[v] = 0.0f;
#pragma unroll
        for (int nt = 0; nt < 4; ++nt) {
#pragma unroll
            for (int v = 0; v < 8; ++v)
                part[v] += silu(acc2[nt][v]) * w3c[nt];
        }
#pragma unroll
        for (int off = 8; off > 0; off >>= 1)
#pragma unroll
            for (int v = 0; v < 8; ++v)
                part[v] += __shfl_xor(part[v], off, 16);

        // lanes with m<8 own row = 8*hf + m
        int row = 8 * hf + m;
        float cutr = __shfl(cut, row & 15, 32);
        float mypart = 0.0f;
#pragma unroll
        for (int v = 0; v < 8; ++v) mypart = (m == v) ? part[v] : mypart;
        if (m < 8) eacc += (mypart + b3v) * cutr;
    }

    // wave-level reduce, one atomic per wave
#pragma unroll
    for (int off = 16; off > 0; off >>= 1) eacc += __shfl_xor(eacc, off, 32);
    if (lane == 0) atomicAdd(out, eacc);
}

// ---------------- kernel 3: per-atom local RBF-by-species features ----------------
__global__ __launch_bounds__(256) void local_feat_kernel(
    const int* __restrict__ species, const float* __restrict__ pos,
    const float* __restrict__ embed, const float* __restrict__ total_charge,
    float* __restrict__ xbuf) {
    __shared__ float acc[NSPEC * NC];
    const int i = blockIdx.x;
    const int tid = threadIdx.x;
    for (int t = tid; t < NSPEC * NC; t += 256) acc[t] = 0.0f;
    __syncthreads();

    const float xi = pos[3 * i + 0], yi = pos[3 * i + 1], zi = pos[3 * i + 2];
    const int lane = tid & 31, wave = tid >> 5;

    for (int j = wave; j < NATOM; j += 8) {
        if (j == i) continue;
        float dx = xi - pos[3 * j + 0];
        float dy = yi - pos[3 * j + 1];
        float dz = zi - pos[3 * j + 2];
        float d = sqrtf(dx * dx + dy * dy + dz * dz + 1e-12f);
        if (d >= CUTR) continue;                  // cutoff -> 0, skip
        float cut = cutoff_fn(d);
        float t = d - (float)lane * (CUTR / 31.0f);   // lane = RBF center index
        float v = __expf(-GAMMA_R * t * t) * cut;
        atomicAdd(&acc[species[j] * NC + lane], v);   // ds_add_f32
    }
    __syncthreads();

    float* xr = xbuf + i * LOCAL_IN;
    xr[tid] = acc[tid];                                    // 256 locals
    if (tid < 32) xr[256 + tid] = embed[species[i] * 32 + tid];
    if (tid == 0) xr[288] = total_charge[0];
}

// ---------------- kernel 4: charge MLP (tiny; scalar FMA) ----------------
__global__ __launch_bounds__(128) void charge_mlp_kernel(
    const float* __restrict__ xbuf,
    const float* __restrict__ cW1, const float* __restrict__ cb1,
    const float* __restrict__ cW2, const float* __restrict__ cb2,
    const float* __restrict__ cW3, const float* __restrict__ cb3,
    const float* __restrict__ charge_scale, float* __restrict__ raw) {
    const int i = blockIdx.x * blockDim.x + threadIdx.x;
    if (i >= NATOM) return;
    const float* xr = xbuf + i * LOCAL_IN;

    float h[HID];
#pragma unroll
    for (int t = 0; t < HID; ++t) h[t] = cb1[t];
    for (int k = 0; k < LOCAL_IN; ++k) {
        float xv = xr[k];
        const float* wr = cW1 + k * HID;
#pragma unroll
        for (int t = 0; t < HID; ++t) h[t] = fmaf(xv, wr[t], h[t]);
    }
#pragma unroll
    for (int t = 0; t < HID; ++t) h[t] = silu(h[t]);

    float h2[HID];
#pragma unroll
    for (int t = 0; t < HID; ++t) h2[t] = cb2[t];
    for (int k = 0; k < HID; ++k) {
        float xv = h[k];
        const float* wr = cW2 + k * HID;
#pragma unroll
        for (int t = 0; t < HID; ++t) h2[t] = fmaf(xv, wr[t], h2[t]);
    }
    float r = cb3[0];
#pragma unroll
    for (int t = 0; t < HID; ++t) r = fmaf(silu(h2[t]), cW3[t], r);
    raw[i] = r * charge_scale[0];
}

// ---------------- kernel 5: mean-shift charges ----------------
__global__ void finalize_kernel(const float* __restrict__ raw,
                                const float* __restrict__ total_charge,
                                float* __restrict__ out, float* __restrict__ q) {
    __shared__ float red[256];
    const int tid = threadIdx.x;
    float s = 0.0f;
    for (int k = tid; k < NATOM; k += 256) s += raw[k];
    red[tid] = s;
    __syncthreads();
    for (int off = 128; off > 0; off >>= 1) {
        if (tid < off) red[tid] += red[tid + off];
        __syncthreads();
    }
    float mean = red[0] * (1.0f / NATOM);
    float add = total_charge[0] * (1.0f / NATOM);
    for (int k = tid; k < NATOM; k += 256) {
        float c = raw[k] - mean + add;
        out[1 + k] = c;
        q[k] = c;
    }
}

// ---------------- kernel 6: long-range Coulomb sum ----------------
__global__ __launch_bounds__(256) void coulomb_kernel(
    const float* __restrict__ pos, const float* __restrict__ q,
    const float* __restrict__ coulomb_scale,
    const float* __restrict__ softening_raw, float* __restrict__ out) {
    float soft = logf(1.0f + __expf(softening_raw[0])) + 0.001f;
    float s2 = soft * soft;
    float acc = 0.0f;
    const int stride = gridDim.x * blockDim.x;
    for (int p = blockIdx.x * blockDim.x + threadIdx.x; p < NPAIR; p += stride) {
        int i, j; pair_ij(p, i, j);
        float dx = pos[3 * i + 0] - pos[3 * j + 0];
        float dy = pos[3 * i + 1] - pos[3 * j + 1];
        float dz = pos[3 * i + 2] - pos[3 * j + 2];
        float d2 = dx * dx + dy * dy + dz * dz + 1e-12f;
        acc += q[i] * q[j] * rsqrtf(d2 + s2);
    }
    acc *= coulomb_scale[0];
#pragma unroll
    for (int off = 16; off > 0; off >>= 1) acc += __shfl_xor(acc, off, 32);
    if ((threadIdx.x & 31) == 0) atomicAdd(out, acc);
}

// ---------------- host launcher ----------------
extern "C" void kernel_launch(void* const* d_in, const int* in_sizes, int n_in,
                              void* d_out, int out_size, void* d_ws, size_t ws_size,
                              hipStream_t stream) {
    (void)in_sizes; (void)n_in; (void)out_size; (void)ws_size;
    const int*   species        = (const int*)d_in[0];
    const float* positions      = (const float*)d_in[1];
    const float* total_charge   = (const float*)d_in[2];
    const float* embed          = (const float*)d_in[3];
    const float* W1             = (const float*)d_in[4];
    const float* b1             = (const float*)d_in[5];
    const float* W2             = (const float*)d_in[6];
    const float* b2             = (const float*)d_in[7];
    const float* W3             = (const float*)d_in[8];
    const float* b3             = (const float*)d_in[9];
    const float* atom_bias      = (const float*)d_in[10];
    const float* cW1            = (const float*)d_in[11];
    const float* cb1            = (const float*)d_in[12];
    const float* cW2            = (const float*)d_in[13];
    const float* cb2            = (const float*)d_in[14];
    const float* cW3            = (const float*)d_in[15];
    const float* cb3            = (const float*)d_in[16];
    const float* charge_scale   = (const float*)d_in[17];
    const float* coulomb_scale  = (const float*)d_in[18];
    const float* softening_raw  = (const float*)d_in[19];

    float* out  = (float*)d_out;            // [0]=energy, [1..NATOM]=charges
    float* xbuf = (float*)d_ws;             // [NATOM][289]
    float* rawb = xbuf + NATOM * LOCAL_IN;  // [NATOM]
    float* qb   = rawb + NATOM;             // [NATOM]

    init_kernel<<<1, 256, 0, stream>>>(species, atom_bias, out);
    pair_energy_kernel<<<512, 256, 0, stream>>>(species, positions, embed,
                                                W1, b1, W2, b2, W3, b3, out);
    local_feat_kernel<<<NATOM, 256, 0, stream>>>(species, positions, embed,
                                                 total_charge, xbuf);
    charge_mlp_kernel<<<(NATOM + 127) / 128, 128, 0, stream>>>(
        xbuf, cW1, cb1, cW2, cb2, cW3, cb3, charge_scale, rawb);
    finalize_kernel<<<1, 256, 0, stream>>>(rawb, total_charge, out, qb);
    coulomb_kernel<<<512, 256, 0, stream>>>(positions, qb, coulomb_scale,
                                            softening_raw, out);
}
